// SparseGCNPredicator_76115410420293
// MI455X (gfx1250) — compile-verified, matrix operand order checked by reading, not run on hardware
//
#include <hip/hip_runtime.h>
#include <hip/hip_bf16.h>

#define NN 100000
#define EE 1600000
#define GG 1024
#define HID 128
#define NF ((long long)NN * HID)

typedef __attribute__((ext_vector_type(16))) __bf16         v16bf;
typedef __attribute__((ext_vector_type(8)))  float          v8f;
typedef __attribute__((ext_vector_type(16))) unsigned short v16us;
typedef __attribute__((ext_vector_type(8)))  unsigned short v8us;

__device__ __forceinline__ unsigned short f2bf(float f) {
    unsigned u = __builtin_bit_cast(unsigned, f);
    u += 0x7FFFu + ((u >> 16) & 1u);   // round-to-nearest-even
    return (unsigned short)(u >> 16);
}

__device__ __forceinline__ void atomAddF(float* p, float v) {
    __hip_atomic_fetch_add(p, v, __ATOMIC_RELAXED, __HIP_MEMORY_SCOPE_AGENT);
}

// ---------------- utility kernels ----------------

__global__ void zero_kernel(float* __restrict__ p, long long n) {
    long long i = (long long)blockIdx.x * blockDim.x + threadIdx.x;
    if (i < n) p[i] = 0.0f;
}

// convert W [K=128][N=128] f32 row-major -> WT bf16 [N][K] (column-major of W)
__global__ void convw_kernel(const float* __restrict__ W, unsigned short* __restrict__ WT) {
    int i = blockIdx.x * blockDim.x + threadIdx.x;     // 16384
    int n = i >> 7, k = i & 127;
    WT[n * 128 + k] = f2bf(W[k * 128 + n]);
}

__global__ void degree_kernel(const int* __restrict__ dst, float* __restrict__ deg, int e) {
    int i = blockIdx.x * blockDim.x + threadIdx.x;
    if (i < e) atomAddF(&deg[dst[i]], 1.0f);
}

__global__ void norm_kernel(float* __restrict__ nrm, int n) {
    int i = blockIdx.x * blockDim.x + threadIdx.x;
    if (i < n) { float d = nrm[i]; nrm[i] = d > 0.0f ? __frsqrt_rn(d) : 0.0f; }
}

// ---------------- WMMA GEMM: Y[M,128] = X[M,128] @ W[128,128] + b, optional relu ----
// WT is bf16, [n][k] layout (W column-major). One block = 16 rows, 8 waves = 8 col tiles.

__global__ void gemm_wmma_kernel(const float* __restrict__ X,
                                 const unsigned short* __restrict__ WT,
                                 const float* __restrict__ bias,
                                 float* __restrict__ Y, int relu) {
    __shared__ unsigned short As[16 * 128];            // 16x128 bf16 A tile
    const int row0 = blockIdx.x * 16;
    const int t    = threadIdx.x;

    // cooperative convert: 256 threads x 8 elements = 2048
    const float* xin = X + (long long)row0 * 128;
    #pragma unroll
    for (int j = 0; j < 8; ++j)
        As[t * 8 + j] = f2bf(xin[t * 8 + j]);
    __syncthreads();

    const int wave = t >> 5;                 // 0..7 -> column tile
    const int lane = t & 31;
    const int l16  = lane & 15;
    const int hi   = lane >> 4;              // 0/1
    const int col  = wave * 16 + l16;        // this lane's N column (B/C/D)

    v8f acc = {};
    #pragma unroll
    for (int kc = 0; kc < 128; kc += 32) {
        const int koff = kc + hi * 8;
        // A fragment: row=l16, K = koff..koff+7 and koff+16..koff+23
        v8us alo = *(const v8us*)&As[l16 * 128 + koff];
        v8us ahi = *(const v8us*)&As[l16 * 128 + koff + 16];
        // B fragment: col fixed, same K pattern; WT row-contiguous in K
        v8us blo = *(const v8us*)&WT[col * 128 + koff];
        v8us bhi = *(const v8us*)&WT[col * 128 + koff + 16];
        v16us au, bu;
        #pragma unroll
        for (int j = 0; j < 8; ++j) {
            au[j] = alo[j]; au[j + 8] = ahi[j];
            bu[j] = blo[j]; bu[j + 8] = bhi[j];
        }
        v16bf a = __builtin_bit_cast(v16bf, au);
        v16bf b = __builtin_bit_cast(v16bf, bu);
        acc = __builtin_amdgcn_wmma_f32_16x16x32_bf16(false, a, false, b,
                                                      (short)0, acc, false, false);
    }

    const float bv = bias[col];
    #pragma unroll
    for (int r = 0; r < 8; ++r) {
        float v = acc[r] + bv;
        if (relu) v = fmaxf(v, 0.0f);
        const int m = row0 + r + hi * 8;     // C/D layout: M = r + 8*hi, N = l16
        Y[(long long)m * 128 + col] = v;
    }
}

// ---------------- edge scatter: agg[dst] += h[src] * norm[src] -------------------
// one wave per edge; lane handles 4 consecutive features (coalesced 512B/edge)

__global__ void edge_scatter_kernel(const float* __restrict__ h,
                                    const float* __restrict__ nrm,
                                    const int* __restrict__ src,
                                    const int* __restrict__ dst,
                                    float* __restrict__ agg) {
    long long tid = (long long)blockIdx.x * blockDim.x + threadIdx.x;
    long long e = tid >> 5;
    if (e >= EE) return;
    int c = (int)(tid & 31) * 4;
    int s = src[e], d = dst[e];
    float ns = nrm[s];
    const float4 hv = *(const float4*)&h[(long long)s * 128 + c];
    float* out = &agg[(long long)d * 128 + c];
    atomAddF(out + 0, hv.x * ns);
    atomAddF(out + 1, hv.y * ns);
    atomAddF(out + 2, hv.z * ns);
    atomAddF(out + 3, hv.w * ns);
}

// out[i] = relu(agg[i] * norm[node])
__global__ void scale_relu_kernel(const float* __restrict__ agg,
                                  const float* __restrict__ nrm,
                                  float* __restrict__ out, long long n) {
    long long i = (long long)blockIdx.x * blockDim.x + threadIdx.x;
    if (i < n) out[i] = fmaxf(agg[i] * nrm[i >> 7], 0.0f);
}

__global__ void count_kernel(const int* __restrict__ gidx, float* __restrict__ cnt, int n) {
    int i = blockIdx.x * blockDim.x + threadIdx.x;
    if (i < n) atomAddF(&cnt[gidx[i]], 1.0f);
}

__global__ void pool_sum_kernel(const float* __restrict__ h,
                                const int* __restrict__ gidx,
                                float* __restrict__ pool, long long n) {
    long long i = (long long)blockIdx.x * blockDim.x + threadIdx.x;
    if (i < n) {
        int node = (int)(i >> 7), f = (int)(i & 127);
        atomAddF(&pool[(long long)gidx[node] * 128 + f], h[i]);
    }
}

__global__ void pool_div_kernel(float* __restrict__ pool, const float* __restrict__ cnt,
                                long long n) {
    long long i = (long long)blockIdx.x * blockDim.x + threadIdx.x;
    if (i < n) pool[i] /= fmaxf(cnt[i >> 7], 1.0f);
}

// final: out[g] = dot(fc[g,:], Wout[:,0]) + bout  (one wave per graph)
__global__ void out_kernel(const float* __restrict__ fc,
                           const float* __restrict__ Wout,
                           const float* __restrict__ bout,
                           float* __restrict__ out) {
    int t = blockIdx.x * blockDim.x + threadIdx.x;
    int g = t >> 5;
    int lane = t & 31;
    if (g >= GG) return;
    const float4 hv = *(const float4*)&fc[(long long)g * 128 + lane * 4];
    const float4 wv = *(const float4*)&Wout[lane * 4];
    float v = hv.x * wv.x + hv.y * wv.y + hv.z * wv.z + hv.w * wv.w;
    #pragma unroll
    for (int off = 16; off > 0; off >>= 1) v += __shfl_xor(v, off, 32);
    if (lane == 0) out[g] = v + bout[0];
}

// ---------------- launch ----------------

extern "C" void kernel_launch(void* const* d_in, const int* in_sizes, int n_in,
                              void* d_out, int out_size, void* d_ws, size_t ws_size,
                              hipStream_t stream) {
    const float* x    = (const float*)d_in[0];
    const int*   adj  = (const int*)d_in[1];
    const int*   gidx = (const int*)d_in[2];
    const float* W1   = (const float*)d_in[4];
    const float* b1   = (const float*)d_in[5];
    const float* W2   = (const float*)d_in[6];
    const float* b2   = (const float*)d_in[7];
    const float* Wfc  = (const float*)d_in[8];
    const float* bfc  = (const float*)d_in[9];
    const float* Wout = (const float*)d_in[10];
    const float* bout = (const float*)d_in[11];
    float* out = (float*)d_out;

    const int* src = adj;
    const int* dst = adj + EE;

    // workspace carve (floats)
    float* bufA = (float*)d_ws;                 // N*128
    float* bufB = bufA + NF;                    // N*128
    float* nrm  = bufB + NF;                    // N
    float* pool = nrm + NN;                     // G*128
    float* cnt  = pool + (long long)GG * 128;   // G
    float* fc   = cnt + GG;                     // G*128
    unsigned short* W1T  = (unsigned short*)(fc + (long long)GG * 128); // 16384 each
    unsigned short* W2T  = W1T + 16384;
    unsigned short* WfcT = W2T + 16384;

    const int B = 256;
    const int rowBlocks = NN / 16;              // 6250, exact
    const long long nfeat = NF;

    // weight convert/transpose to bf16
    convw_kernel<<<16384 / B, B, 0, stream>>>(W1, W1T);
    convw_kernel<<<16384 / B, B, 0, stream>>>(W2, W2T);
    convw_kernel<<<16384 / B, B, 0, stream>>>(Wfc, WfcT);

    // degrees -> norm
    zero_kernel<<<(NN + B - 1) / B, B, 0, stream>>>(nrm, NN);
    degree_kernel<<<(EE + B - 1) / B, B, 0, stream>>>(dst, nrm, EE);
    norm_kernel<<<(NN + B - 1) / B, B, 0, stream>>>(nrm, NN);

    // layer 1: h1 = x@W1 + b1 -> bufA; agg -> bufB; relu(agg*norm) -> bufA
    gemm_wmma_kernel<<<rowBlocks, B, 0, stream>>>(x, W1T, b1, bufA, 0);
    zero_kernel<<<(int)((nfeat + B - 1) / B), B, 0, stream>>>(bufB, nfeat);
    edge_scatter_kernel<<<(int)(((long long)EE * 32 + B - 1) / B), B, 0, stream>>>(bufA, nrm, src, dst, bufB);
    scale_relu_kernel<<<(int)((nfeat + B - 1) / B), B, 0, stream>>>(bufB, nrm, bufA, nfeat);

    // layer 2: h2 = bufA@W2 + b2 -> bufB; agg -> bufA; relu -> bufB
    gemm_wmma_kernel<<<rowBlocks, B, 0, stream>>>(bufA, W2T, b2, bufB, 0);
    zero_kernel<<<(int)((nfeat + B - 1) / B), B, 0, stream>>>(bufA, nfeat);
    edge_scatter_kernel<<<(int)(((long long)EE * 32 + B - 1) / B), B, 0, stream>>>(bufB, nrm, src, dst, bufA);
    scale_relu_kernel<<<(int)((nfeat + B - 1) / B), B, 0, stream>>>(bufA, nrm, bufB, nfeat);

    // segment mean pool -> pool[G,128]
    const long long gf = (long long)GG * 128;
    zero_kernel<<<(int)((gf + B - 1) / B), B, 0, stream>>>(pool, gf);
    zero_kernel<<<(GG + B - 1) / B, B, 0, stream>>>(cnt, GG);
    count_kernel<<<(NN + B - 1) / B, B, 0, stream>>>(gidx, cnt, NN);
    pool_sum_kernel<<<(int)((nfeat + B - 1) / B), B, 0, stream>>>(bufB, gidx, pool, nfeat);
    pool_div_kernel<<<(int)((gf + B - 1) / B), B, 0, stream>>>(pool, cnt, gf);

    // MLP head: relu(pool@Wfc + bfc) -> fc; out = fc@Wout + bout
    gemm_wmma_kernel<<<GG / 16, B, 0, stream>>>(pool, WfcT, bfc, fc, 1);
    out_kernel<<<(GG * 32 + B - 1) / B, B, 0, stream>>>(fc, Wout, bout, out);
}
